// EstraNetBlock_67138928771247
// MI455X (gfx1250) — matrix operand order repaired and checked by reading, not compiled, not run on hardware
//
#include <hip/hip_runtime.h>
#include <hip/hip_bf16.h>

// ---------------------------------------------------------------------------
// CDNA5 (gfx1250) EstraNet block: linear-attention transformer layer.
// GEMMs on v_wmma_f32_16x16x32_bf16; tile staging via GLOBAL_LOAD_ASYNC_TO_LDS
// (ASYNCcnt-tracked, double-buffered) when the builtins are available.
// ---------------------------------------------------------------------------

typedef __attribute__((ext_vector_type(16))) __bf16       bf16x16;
typedef __attribute__((ext_vector_type(8)))  float        f32x8;
typedef __attribute__((ext_vector_type(4)))  unsigned int u32x4;

// exact parameter type of the async-LDS builtins (per hipcc diagnostic):
// '__attribute__((__vector_size__(4 * sizeof(int)))) int *' in AS1/AS3
typedef int v4i_gcc __attribute__((vector_size(4 * sizeof(int))));
typedef __attribute__((address_space(1))) v4i_gcc* gptr_v4i;
typedef __attribute__((address_space(3))) v4i_gcc* lptr_v4i;

union FragU { u32x4 u[2]; bf16x16 v; };            // 32B WMMA A/B fragment
union VecU  { u32x4 u;    unsigned short s[8]; };  // 16B = 8 bf16

#if defined(__has_builtin)
#if __has_builtin(__builtin_amdgcn_global_load_async_to_lds_b128) && \
    __has_builtin(__builtin_amdgcn_s_wait_asynccnt)
#define USE_ASYNC_LDS 1
#endif
#endif
#ifndef USE_ASYNC_LDS
#define USE_ASYNC_LDS 0
#endif

#if USE_ASYNC_LDS
__device__ __forceinline__ void async_copy16(const unsigned short* g,
                                             unsigned short* l) {
    __builtin_amdgcn_global_load_async_to_lds_b128(
        (gptr_v4i)(void*)g, (lptr_v4i)(void*)l, 0, 0);
}
#endif

__device__ __forceinline__ unsigned short f2bf(float f) {
    unsigned int u = __float_as_uint(f);
    // round-to-nearest-even bf16 truncation
    unsigned int r = (u + 0x7FFFu + ((u >> 16) & 1u)) >> 16;
    return (unsigned short)r;
}

// ---------------------------------------------------------------------------
// Generic tiled WMMA GEMM:  C = A(MxK, bf16) * B(KxN, bf16)  [+bias][relu]
// OUT_F32: write f32 (+optional f32 residual).  FOURIER: write
// cos(p/8)/sqrt(128) at col n and sin(p/8)/sqrt(128) at col n+ldc/2 (bf16).
// Batched via blockIdx.z with (outer,inner) strides, z = outer*nInner+inner.
// ---------------------------------------------------------------------------
template<int BM, int BN, int WM, int WN, bool OUT_F32, bool FOURIER>
__global__ __launch_bounds__((BM / WM) * (BN / WN) * 32)
void gemm_bf16_kernel(const unsigned short* __restrict__ A,
                      const unsigned short* __restrict__ B,
                      void* __restrict__ Cv,
                      const float* __restrict__ bias,
                      const float* __restrict__ resid,
                      int M, int N, int K,
                      long lda, long ldb, long ldc,
                      long aOuter, long aInner, long bOuter, long bInner,
                      long cOuter, long cInner, int nInner, int relu)
{
    constexpr int BK = 32;
    constexpr int PA = BK + 8;           // bf16 pitch -> 80B rows (16B aligned)
    constexpr int PB = BN + 8;
    constexpr int NT = (BM / WM) * (BN / WN) * 32;
    constexpr int TM = WM / 16;
    constexpr int TN = WN / 16;
    constexpr int ITA = (BM * BK / 8) / NT;   // 16B chunks per thread (A)
    constexpr int ITB = (BK * BN / 8) / NT;   // 16B chunks per thread (B)
#if USE_ASYNC_LDS
    constexpr int NBUF = 2;
    constexpr int NASY = ITA + ITB;           // async instrs per wave per tile
#else
    constexpr int NBUF = 1;
#endif

    __shared__ __align__(16) unsigned short sA[NBUF][BM * PA];
    __shared__ __align__(16) unsigned short sB[NBUF][BK * PB];

    const int tid  = threadIdx.x;
    const int lane = tid & 31;
    const int hh   = lane >> 4;          // lane-half select
    const int lm   = lane & 15;
    const int wave = tid >> 5;
    const int wRow = wave / (BN / WN);
    const int wCol = wave % (BN / WN);

    const int  z     = blockIdx.z;
    const long aBase = aOuter * (long)(z / nInner) + aInner * (long)(z % nInner);
    const long bBase = bOuter * (long)(z / nInner) + bInner * (long)(z % nInner);
    const long cBase = cOuter * (long)(z / nInner) + cInner * (long)(z % nInner);

    const long rowBlock = (long)blockIdx.y * BM;
    const long colBlock = (long)blockIdx.x * BN;

    f32x8 acc[TM][TN];
#pragma unroll
    for (int i = 0; i < TM; ++i)
#pragma unroll
        for (int j = 0; j < TN; ++j)
#pragma unroll
            for (int e = 0; e < 8; ++e) acc[i][j][e] = 0.0f;

#if USE_ASYNC_LDS
    // Issue async global->LDS copies for tile at k0 into buffer `buf`.
    auto issue_tile = [&](int k0, int buf) {
#pragma unroll
        for (int i = 0; i < ITA; ++i) {
            int c  = tid + i * NT;
            int r  = c / (BK / 8);
            int cc = (c % (BK / 8)) * 8;
            async_copy16(A + aBase + (rowBlock + r) * lda + (k0 + cc),
                         &sA[buf][r * PA + cc]);
        }
#pragma unroll
        for (int i = 0; i < ITB; ++i) {
            int c  = tid + i * NT;
            int r  = c / (BN / 8);
            int cc = (c % (BN / 8)) * 8;
            async_copy16(B + bBase + (long)(k0 + r) * ldb + (colBlock + cc),
                         &sB[buf][r * PB + cc]);
        }
    };

    issue_tile(0, 0);
    int buf = 0;
#endif

    for (int k0 = 0; k0 < K; k0 += BK) {
#if USE_ASYNC_LDS
        const bool hasNext = (k0 + BK) < K;
        if (hasNext) {
            issue_tile(k0 + BK, buf ^ 1);            // prefetch next (overlaps)
            __builtin_amdgcn_s_wait_asynccnt(NASY);  // current tile landed
        } else {
            __builtin_amdgcn_s_wait_asynccnt(0);
        }
        __syncthreads();
#else
        constexpr int buf = 0;
        // ---- synchronous staging fallback ----
#pragma unroll
        for (int i = 0; i < ITA; ++i) {
            int c  = tid + i * NT;
            int r  = c / (BK / 8);
            int cc = (c % (BK / 8)) * 8;
            u32x4 val = *(const u32x4*)(A + aBase + (rowBlock + r) * lda + (k0 + cc));
            *(u32x4*)(&sA[0][r * PA + cc]) = val;
        }
#pragma unroll
        for (int i = 0; i < ITB; ++i) {
            int c  = tid + i * NT;
            int r  = c / (BN / 8);
            int cc = (c % (BN / 8)) * 8;
            u32x4 val = *(const u32x4*)(B + bBase + (long)(k0 + r) * ldb + (colBlock + cc));
            *(u32x4*)(&sB[0][r * PB + cc]) = val;
        }
        __syncthreads();
#endif

        // ---- fragments per CDNA5 16-bit A(16x32)/B(32x16) VGPR layouts ----
        FragU aF[TM];
        FragU bF[TN];
#pragma unroll
        for (int tr = 0; tr < TM; ++tr) {
            int m = wRow * WM + tr * 16 + lm;
            aF[tr].u[0] = *(const u32x4*)(&sA[buf][m * PA + hh * 8]);      // K 0..15
            aF[tr].u[1] = *(const u32x4*)(&sA[buf][m * PA + 16 + hh * 8]); // K 16..31
        }
#pragma unroll
        for (int tc = 0; tc < TN; ++tc) {
            int n0 = wCol * WN + tc * 16;
#pragma unroll
            for (int s = 0; s < 16; ++s) {
                int kk = (s < 8) ? (hh * 8 + s) : (16 + hh * 8 + (s - 8));
                ((unsigned short*)&bF[tc])[s] = sB[buf][kk * PB + n0 + lm];
            }
        }
#pragma unroll
        for (int tr = 0; tr < TM; ++tr)
#pragma unroll
            for (int tc = 0; tc < TN; ++tc)
                acc[tr][tc] = __builtin_amdgcn_wmma_f32_16x16x32_bf16(
                    false, aF[tr].v, false, bF[tc].v, (short)0, acc[tr][tc],
                    false, false);
        __syncthreads();
#if USE_ASYNC_LDS
        buf ^= 1;
#endif
    }

    // ---- epilogue (C layout: VGPR r -> row r + 8*half, col = lane&15) ----
#pragma unroll
    for (int tr = 0; tr < TM; ++tr) {
#pragma unroll
        for (int tc = 0; tc < TN; ++tc) {
            long row0 = rowBlock + wRow * WM + tr * 16;
            long n    = colBlock + wCol * WN + tc * 16 + lm;
#pragma unroll
            for (int r = 0; r < 8; ++r) {
                long  row = row0 + r + 8 * hh;
                float v   = acc[tr][tc][r];
                if (FOURIER) {
                    float p = v * 0.125f;                       // / sqrt(d_head)
                    unsigned short* C = (unsigned short*)Cv;
                    const float sc = 0.08838834764831845f;      // 1/sqrt(128)
                    C[cBase + row * ldc + n]              = f2bf(__cosf(p) * sc);
                    C[cBase + row * ldc + n + (ldc >> 1)] = f2bf(__sinf(p) * sc);
                } else {
                    if (bias)  v += bias[n];
                    if (relu)  v = v > 0.0f ? v : 0.0f;
                    if (OUT_F32) {
                        float o = v;
                        if (resid) o += resid[cBase + row * ldc + n];
                        ((float*)Cv)[cBase + row * ldc + n] = o;
                    } else {
                        ((unsigned short*)Cv)[cBase + row * ldc + n] = f2bf(v);
                    }
                }
            }
        }
    }
}

// ---------------------------------------------------------------------------
// kv = kp^T * v  :  C[256,64] (bf16) = sum_k A[k,m] * B[k,n], A staged
// transposed through LDS. One workgroup per (batch, head); K-loop over L.
// ---------------------------------------------------------------------------
__global__ __launch_bounds__(256)
void gemm_at_kernel(const unsigned short* __restrict__ A,
                    const unsigned short* __restrict__ B,
                    unsigned short* __restrict__ C,
                    int K, long lda, long ldb,
                    long aOuter, long aInner, long bOuter, long bInner,
                    long cStride, int nInner)
{
    constexpr int M = 256, N = 64, BK = 32;
    constexpr int PA = BK + 8;   // 40
    constexpr int PB = N + 8;    // 72
    constexpr int WM = 64, WN = 32, TM = 4, TN = 2;

    __shared__ __align__(16) unsigned short sAt[M * PA];   // 20 KB, transposed A
    __shared__ __align__(16) unsigned short sBt[BK * PB];  // 4.5 KB

    const int tid  = threadIdx.x;
    const int lane = tid & 31;
    const int hh   = lane >> 4;
    const int lm   = lane & 15;
    const int wave = tid >> 5;
    const int wRow = wave >> 1;
    const int wCol = wave & 1;

    const int  z     = blockIdx.x;
    const long aBase = aOuter * (long)(z / nInner) + aInner * (long)(z % nInner);
    const long bBase = bOuter * (long)(z / nInner) + bInner * (long)(z % nInner);
    const long cBase = cStride * (long)z;

    f32x8 acc[TM][TN];
#pragma unroll
    for (int i = 0; i < TM; ++i)
#pragma unroll
        for (int j = 0; j < TN; ++j)
#pragma unroll
            for (int e = 0; e < 8; ++e) acc[i][j][e] = 0.0f;

    for (int k0 = 0; k0 < K; k0 += BK) {
        // stage A [BK x 256] transposed -> sAt[m][k]
#pragma unroll
        for (int i = 0; i < 4; ++i) {
            int c  = tid + i * 256;          // 0..1023
            int k  = c >> 5;                 // 0..31
            int m0 = (c & 31) * 8;           // 0..248
            VecU v;
            v.u = *(const u32x4*)(A + aBase + (long)(k0 + k) * lda + m0);
#pragma unroll
            for (int j = 0; j < 8; ++j) sAt[(m0 + j) * PA + k] = v.s[j];
        }
        // stage B [BK x 64]
        {
            int k  = tid >> 3;
            int n0 = (tid & 7) * 8;
            u32x4 v = *(const u32x4*)(B + bBase + (long)(k0 + k) * ldb + n0);
            *(u32x4*)(sBt + k * PB + n0) = v;
        }
        __syncthreads();

        FragU aF[TM];
        FragU bF[TN];
#pragma unroll
        for (int tr = 0; tr < TM; ++tr) {
            int m = wRow * WM + tr * 16 + lm;
            aF[tr].u[0] = *(const u32x4*)(sAt + m * PA + hh * 8);
            aF[tr].u[1] = *(const u32x4*)(sAt + m * PA + 16 + hh * 8);
        }
#pragma unroll
        for (int tc = 0; tc < TN; ++tc) {
            int n0 = wCol * WN + tc * 16;
#pragma unroll
            for (int s = 0; s < 16; ++s) {
                int kk = (s < 8) ? (hh * 8 + s) : (16 + hh * 8 + (s - 8));
                ((unsigned short*)&bF[tc])[s] = sBt[kk * PB + n0 + lm];
            }
        }
#pragma unroll
        for (int tr = 0; tr < TM; ++tr)
#pragma unroll
            for (int tc = 0; tc < TN; ++tc)
                acc[tr][tc] = __builtin_amdgcn_wmma_f32_16x16x32_bf16(
                    false, aF[tr].v, false, bF[tc].v, (short)0, acc[tr][tc],
                    false, false);
        __syncthreads();
    }

#pragma unroll
    for (int tr = 0; tr < TM; ++tr)
#pragma unroll
        for (int tc = 0; tc < TN; ++tc) {
            int row0 = wRow * WM + tr * 16;
            int n    = wCol * WN + tc * 16 + lm;
#pragma unroll
            for (int r = 0; r < 8; ++r) {
                int row = row0 + r + 8 * hh;
                C[cBase + (long)row * N + n] = f2bf(acc[tr][tc][r]);
            }
        }
}

// ---------------------------------------------------------------------------
// LayerNorm over D=1024, one token per 256-thread block, bf16 output.
// ---------------------------------------------------------------------------
__global__ __launch_bounds__(256)
void ln_kernel(const float* __restrict__ x, const float* __restrict__ g,
               const float* __restrict__ b, unsigned short* __restrict__ out)
{
    const long t   = blockIdx.x;
    const int  tid = threadIdx.x;
    const float* row = x + t * 1024;

    float v[4];
    float s = 0.0f, ss = 0.0f;
#pragma unroll
    for (int j = 0; j < 4; ++j) {
        int c = tid + j * 256;
        v[j] = row[c];
        s  += v[j];
        ss += v[j] * v[j];
    }
    __shared__ float rs[256];
    __shared__ float rq[256];
    rs[tid] = s;
    rq[tid] = ss;
    __syncthreads();
    for (int st = 128; st > 0; st >>= 1) {
        if (tid < st) { rs[tid] += rs[tid + st]; rq[tid] += rq[tid + st]; }
        __syncthreads();
    }
    float mu   = rs[0] * (1.0f / 1024.0f);
    float var  = rq[0] * (1.0f / 1024.0f) - mu * mu;
    float rstd = rsqrtf(var + 1e-5f);
#pragma unroll
    for (int j = 0; j < 4; ++j) {
        int c = tid + j * 256;
        out[t * 1024 + c] = f2bf((v[j] - mu) * rstd * g[c] + b[c]);
    }
}

// ---------------------------------------------------------------------------
// f32 -> bf16 conversion (weights)
// ---------------------------------------------------------------------------
__global__ void cvt_kernel(const float* __restrict__ in,
                           unsigned short* __restrict__ out, long n)
{
    long i      = (long)blockIdx.x * blockDim.x + threadIdx.x;
    long stride = (long)gridDim.x * blockDim.x;
    for (; i < n; i += stride) out[i] = f2bf(in[i]);
}

// ---------------------------------------------------------------------------
// Host orchestration
// ---------------------------------------------------------------------------
extern "C" void kernel_launch(void* const* d_in, const int* in_sizes, int n_in,
                              void* d_out, int out_size, void* d_ws, size_t ws_size,
                              hipStream_t stream)
{
    (void)in_sizes; (void)n_in; (void)out_size; (void)ws_size;

    const float* x    = (const float*)d_in[0];
    const float* proj = (const float*)d_in[1];
    const float* wq   = (const float*)d_in[2];
    const float* bq   = (const float*)d_in[3];
    const float* wk   = (const float*)d_in[4];
    const float* bk   = (const float*)d_in[5];
    const float* wv   = (const float*)d_in[6];
    const float* bv   = (const float*)d_in[7];
    const float* wo   = (const float*)d_in[8];
    const float* bo   = (const float*)d_in[9];
    const float* ln1g = (const float*)d_in[10];
    const float* ln1b = (const float*)d_in[11];
    const float* w1   = (const float*)d_in[12];
    const float* b1   = (const float*)d_in[13];
    const float* w2   = (const float*)d_in[14];
    const float* b2   = (const float*)d_in[15];
    const float* ln2g = (const float*)d_in[16];
    const float* ln2b = (const float*)d_in[17];

    const long T = 16384;            // B*L
    const long D = 1024;

    // Workspace bump allocator (256B aligned)
    char* wp = (char*)d_ws;
    auto alloc = [&](size_t bytes) -> void* {
        void* p = (void*)wp;
        wp += (bytes + 255) & ~(size_t)255;
        return p;
    };
    unsigned short* x2b   = (unsigned short*)alloc(T * D * 2);
    unsigned short* qb    = (unsigned short*)alloc(T * D * 2);
    unsigned short* kb    = (unsigned short*)alloc(T * D * 2);
    unsigned short* vb    = (unsigned short*)alloc(T * D * 2);
    unsigned short* qpb   = (unsigned short*)alloc(T * 16 * 256 * 2);
    unsigned short* kpb   = (unsigned short*)alloc(T * 16 * 256 * 2);
    unsigned short* kvb   = (unsigned short*)alloc(64L * 256 * 64 * 2);
    unsigned short* attnb = (unsigned short*)alloc(T * D * 2);
    unsigned short* y2b   = (unsigned short*)alloc(T * D * 2);
    unsigned short* hb    = (unsigned short*)alloc(T * 4096 * 2);
    unsigned short* wqb   = (unsigned short*)alloc(D * D * 2);
    unsigned short* wkb   = (unsigned short*)alloc(D * D * 2);
    unsigned short* wvb   = (unsigned short*)alloc(D * D * 2);
    unsigned short* wob   = (unsigned short*)alloc(D * D * 2);
    unsigned short* w1b   = (unsigned short*)alloc(D * 4096 * 2);
    unsigned short* w2b   = (unsigned short*)alloc(4096 * D * 2);
    unsigned short* projb = (unsigned short*)alloc(64 * 128 * 2);

    auto cvt = [&](const float* in, unsigned short* out, long n) {
        int blocks = (int)((n + 1023) / 1024);
        cvt_kernel<<<blocks, 256, 0, stream>>>(in, out, n);
    };
    cvt(wq, wqb, D * D);
    cvt(wk, wkb, D * D);
    cvt(wv, wvb, D * D);
    cvt(wo, wob, D * D);
    cvt(w1, w1b, D * 4096);
    cvt(w2, w2b, 4096 * D);
    cvt(proj, projb, 64 * 128);

    // ---- LN1 ----
    ln_kernel<<<(int)T, 256, 0, stream>>>(x, ln1g, ln1b, x2b);

    // ---- Q, K, V projections: [16384,1024] x [1024,1024] + bias -> bf16 ----
    dim3 gQKV(1024 / 128, (unsigned)(T / 128), 1);
    gemm_bf16_kernel<128, 128, 32, 64, false, false><<<gQKV, 256, 0, stream>>>(
        x2b, wqb, qb, bq, nullptr, (int)T, 1024, 1024, 1024, 1024, 1024,
        0, 0, 0, 0, 0, 0, 1, 0);
    gemm_bf16_kernel<128, 128, 32, 64, false, false><<<gQKV, 256, 0, stream>>>(
        x2b, wkb, kb, bk, nullptr, (int)T, 1024, 1024, 1024, 1024, 1024,
        0, 0, 0, 0, 0, 0, 1, 0);
    gemm_bf16_kernel<128, 128, 32, 64, false, false><<<gQKV, 256, 0, stream>>>(
        x2b, wvb, vb, bv, nullptr, (int)T, 1024, 1024, 1024, 1024, 1024,
        0, 0, 0, 0, 0, 0, 1, 0);

    // ---- Fourier features: [262144,64] x [64,128], fused cos/sin epilogue ----
    dim3 gF(1, (unsigned)(T * 16 / 128), 1);
    gemm_bf16_kernel<128, 128, 32, 64, false, true><<<gF, 256, 0, stream>>>(
        qb, projb, qpb, nullptr, nullptr, (int)(T * 16), 128, 64, 64, 128, 256,
        0, 0, 0, 0, 0, 0, 1, 0);
    gemm_bf16_kernel<128, 128, 32, 64, false, true><<<gF, 256, 0, stream>>>(
        kb, projb, kpb, nullptr, nullptr, (int)(T * 16), 128, 64, 64, 128, 256,
        0, 0, 0, 0, 0, 0, 1, 0);

    // ---- kv[b,h] = kp^T v : 64 x (256 x 4096 x 64) ----
    gemm_at_kernel<<<64, 256, 0, stream>>>(
        kpb, vb, kvb, 4096, 4096, 1024,
        16777216L, 256L,      // kp: per-b / per-h offsets
        4194304L, 64L,        // v:  per-b / per-h offsets
        16384L, 16);

    // ---- attn[b,:,h,:] = qp[b,:,h,:] @ kv[b,h] : batched 4096x256x64 ----
    dim3 gA(1, 4096 / 128, 64);
    gemm_bf16_kernel<128, 64, 32, 32, false, false><<<gA, 256, 0, stream>>>(
        qpb, kvb, attnb, nullptr, nullptr, 4096, 64, 256,
        4096, 64, 1024,
        16777216L, 256L,      // qp batch offsets
        262144L, 16384L,      // kv batch offsets
        4194304L, 64L,        // attn batch offsets
        16, 0);

    // ---- wo projection + residual -> d_out (f32) ----
    dim3 gO(1024 / 128, (unsigned)(T / 128), 1);
    gemm_bf16_kernel<128, 128, 32, 64, true, false><<<gO, 256, 0, stream>>>(
        attnb, wob, d_out, bo, x, (int)T, 1024, 1024, 1024, 1024, 1024,
        0, 0, 0, 0, 0, 0, 1, 0);

    // ---- LN2 ----
    ln_kernel<<<(int)T, 256, 0, stream>>>((const float*)d_out, ln2g, ln2b, y2b);

    // ---- FFN up + ReLU -> bf16 ----
    dim3 gU(4096 / 128, (unsigned)(T / 128), 1);
    gemm_bf16_kernel<128, 128, 32, 64, false, false><<<gU, 256, 0, stream>>>(
        y2b, w1b, hb, b1, nullptr, (int)T, 4096, 1024, 1024, 4096, 4096,
        0, 0, 0, 0, 0, 0, 1, 1);

    // ---- FFN down + residual (in-place on d_out) ----
    dim3 gD(1024 / 128, (unsigned)(T / 128), 1);
    gemm_bf16_kernel<128, 128, 32, 64, true, false><<<gD, 256, 0, stream>>>(
        hb, w2b, d_out, b2, (const float*)d_out, (int)T, 1024, 4096,
        4096, 1024, 1024, 0, 0, 0, 0, 0, 0, 1, 0);
}
